// MLBAttention_50989851738892
// MI455X (gfx1250) — matrix operand order repaired and checked by reading, not compiled, not run on hardware
//
#include <hip/hip_runtime.h>
#include <hip/hip_bf16.h>
#include <math.h>

// ---------------------------------------------------------------------------
// MLB attention, fused for MI455X (gfx1250, wave32, WMMA + async-to-LDS).
//   B=32, A=4096, F=512, H=512, C=512
// ---------------------------------------------------------------------------

#define B_ 32
#define A_ 4096
#define F_ 512
#define H_ 512
#define C_ 512
#define NEGBIG (-1e30f)

typedef __attribute__((ext_vector_type(16))) __bf16 v16bf;
typedef __attribute__((ext_vector_type(8)))  float  v8f;
typedef __attribute__((ext_vector_type(4)))  float  v4f;

static __device__ __forceinline__ float tanh_fast(float x) {
    // tanh(x) = 1 - 2/(exp(2x)+1); saturates correctly at +/-inf.
    float e = __expf(2.0f * x);
    return 1.0f - 2.0f / (e + 1.0f);
}

// ---------------------------------------------------------------------------
// 1) W_ih (f32 [H][F]) -> bf16 same layout (native cvt)
// ---------------------------------------------------------------------------
__global__ __launch_bounds__(256) void k_cvt_w(const float* __restrict__ w,
                                               unsigned short* __restrict__ wb) {
    int i = blockIdx.x * 256 + threadIdx.x;   // grid covers H_*F_ exactly
    wb[i] = __builtin_bit_cast(unsigned short, (__bf16)w[i]);
}

// ---------------------------------------------------------------------------
// 2) v[b][h] = tanh(context[b] . W_ch[h]) * W_s[0][h]
// ---------------------------------------------------------------------------
__global__ __launch_bounds__(512) void k_ctx(const float* __restrict__ ctx,
                                             const float* __restrict__ Wch,
                                             const float* __restrict__ Ws,
                                             float* __restrict__ v) {
    __shared__ float c[C_];
    int b = blockIdx.x;
    int h = threadIdx.x;                      // 512 threads == H_
    c[h] = ctx[b * C_ + h];
    __syncthreads();
    const v4f* wrow = (const v4f*)(Wch + (size_t)h * C_);
    const v4f* cc   = (const v4f*)c;
    float acc = 0.0f;
#pragma unroll 4
    for (int k = 0; k < C_ / 4; ++k) {
        v4f a = wrow[k];
        v4f b4 = cc[k];
        acc += a.x * b4.x + a.y * b4.y + a.z * b4.z + a.w * b4.w;
    }
    v[b * H_ + h] = tanh_fast(acc) * Ws[h];
}

// ---------------------------------------------------------------------------
// 3) scores[b][a] = sum_h tanh( (inputs @ W_ih^T)[b,a,h] ) * v[b][h]
//    One wave per 16-row tile; A-fragments (16x512 bf16) held in VGPRs.
//    Weight slabs (16 h-rows x 512 K bf16 = 16KB) double-buffered in LDS via
//    GLOBAL_LOAD_ASYNC_TO_LDS_B128 (ASYNCcnt), overlapping the WMMA chain.
// ---------------------------------------------------------------------------
__global__ __launch_bounds__(256) void k_scores(const float* __restrict__ inputs,
                                                const unsigned short* __restrict__ wb,
                                                const float* __restrict__ v,
                                                float* __restrict__ scores) {
    __shared__ unsigned short ldsw[2][16 * F_];        // 2 x 16 KB

    const int tid  = threadIdx.x;
    const int wave = tid >> 5;
    const int lane = tid & 31;
    const int n    = lane & 15;                        // M for A-frag, N for B-frag
    const int half = lane >> 4;

    const int tile    = blockIdx.x * 8 + wave;         // 8192 tiles total
    const int rowbase = tile * 16;
    const int b       = rowbase >> 12;                 // rowbase / A_

    // 32-bit LDS byte offsets of the two staging buffers (addrspace(3))
    typedef __attribute__((address_space(3))) unsigned short lds_us;
    const unsigned lbuf0 = (unsigned)(size_t)(lds_us*)&ldsw[0][0] + tid * 16u;
    const unsigned lbuf1 = (unsigned)(size_t)(lds_us*)&ldsw[1][0] + tid * 16u;

    // Issue the async DMA of weight tile `ct` into LDS buffer at `laddr`:
    // 256 threads x 4 x 16B = 16 KB, per-lane global->LDS, no VGPR data path.
    auto stage = [&](int ct, unsigned laddr) {
        unsigned long long ga = (unsigned long long)(size_t)
            ((const char*)(wb + (size_t)(ct * 16) * F_) + tid * 16);
        asm volatile("global_load_async_to_lds_b128 %0, %1, off"
                     :: "v"(laddr), "v"(ga) : "memory");
        asm volatile("global_load_async_to_lds_b128 %0, %1, off offset:4096"
                     :: "v"(laddr), "v"(ga) : "memory");
        asm volatile("global_load_async_to_lds_b128 %0, %1, off offset:8192"
                     :: "v"(laddr), "v"(ga) : "memory");
        asm volatile("global_load_async_to_lds_b128 %0, %1, off offset:12288"
                     :: "v"(laddr), "v"(ga) : "memory");
    };

    // ---- load & convert A fragments: 16 K-steps of 16x32 bf16 --------------
    // A layout matched to B: lane (m = lane&15, half = lane>>4) holds, for
    // K-block kk, the 16 contiguous elements at K = half*16 .. half*16+15;
    // fragment elements are exactly [chunk0, chunk1].
    const float* arow = inputs + ((size_t)(rowbase + n)) * F_;
    v16bf afrag[16];
#pragma unroll
    for (int kk = 0; kk < 16; ++kk) {
        const float* p0 = arow + kk * 32 + half * 16;
        const float* p1 = p0 + 8;
        v4f c0 = *(const v4f*)(p0);
        v4f c1 = *(const v4f*)(p0 + 4);
        v4f c2 = *(const v4f*)(p1);
        v4f c3 = *(const v4f*)(p1 + 4);
        v16bf fr;
        fr[0]  = (__bf16)c0.x; fr[1]  = (__bf16)c0.y;
        fr[2]  = (__bf16)c0.z; fr[3]  = (__bf16)c0.w;
        fr[4]  = (__bf16)c1.x; fr[5]  = (__bf16)c1.y;
        fr[6]  = (__bf16)c1.z; fr[7]  = (__bf16)c1.w;
        fr[8]  = (__bf16)c2.x; fr[9]  = (__bf16)c2.y;
        fr[10] = (__bf16)c2.z; fr[11] = (__bf16)c2.w;
        fr[12] = (__bf16)c3.x; fr[13] = (__bf16)c3.y;
        fr[14] = (__bf16)c3.z; fr[15] = (__bf16)c3.w;
        afrag[kk] = fr;
    }

    float part[8];
#pragma unroll
    for (int d = 0; d < 8; ++d) part[d] = 0.0f;

    // ---- prologue: stage ct=0 ---------------------------------------------
    stage(0, lbuf0);
    asm volatile("s_wait_asynccnt 0x0" ::: "memory");
    __syncthreads();

    // ---- loop over 32 column tiles (h blocks of 16), double buffered ------
    for (int ct = 0; ct < 32; ++ct) {
        const int cur = ct & 1;
        if (ct < 31)
            stage(ct + 1, cur ? lbuf0 : lbuf1);        // fill spare buffer

        const unsigned short* wl = &ldsw[cur][0];
        v8f acc = {0.f, 0.f, 0.f, 0.f, 0.f, 0.f, 0.f, 0.f};
#pragma unroll
        for (int kk = 0; kk < 16; ++kk) {
            // B-fragment: lane (N=n, half) takes 16 contiguous K values of
            // weight row h = ct*16+n at K = kk*32 + half*16  (32B ds_load)
            const v16bf bfrag =
                *(const v16bf*)(wl + (size_t)n * F_ + kk * 32 + half * 16);
            acc = __builtin_amdgcn_wmma_f32_16x16x32_bf16(
                false, afrag[kk], false, bfrag, (short)0, acc, false, false);
        }

        const float vv = v[b * H_ + ct * 16 + n];      // folded c*W_s column
#pragma unroll
        for (int d = 0; d < 8; ++d)
            part[d] += tanh_fast(acc[d]) * vv;

        asm volatile("s_wait_asynccnt 0x0" ::: "memory");
        __syncthreads();
    }

    // ---- reduce over the 16 lanes of each half (N dimension) --------------
#pragma unroll
    for (int d = 0; d < 8; ++d) {
        float p = part[d];
        p += __shfl_xor(p, 1);
        p += __shfl_xor(p, 2);
        p += __shfl_xor(p, 4);
        p += __shfl_xor(p, 8);
        part[d] = p;
    }
    if (n == 0) {
        // lanes 0..15 hold rows 0..7, lanes 16..31 hold rows 8..15
#pragma unroll
        for (int d = 0; d < 8; ++d)
            scores[rowbase + half * 8 + d] = part[d];
    }
}

// ---------------------------------------------------------------------------
// 4) masked softmax over A per batch (deterministic fixed-order reductions)
// ---------------------------------------------------------------------------
__global__ __launch_bounds__(256) void k_softmax(const float* __restrict__ scores,
                                                 const unsigned char* __restrict__ mask,
                                                 float* __restrict__ attn) {
    __shared__ float redm[8];
    __shared__ float reds[8];
    const int b    = blockIdx.x;
    const int tid  = threadIdx.x;
    const int wave = tid >> 5;
    const int lane = tid & 31;

    const float* s = scores + (size_t)b * A_;
    const unsigned char* m = mask + (size_t)b * A_;

    float vals[16];
    float lmax = -INFINITY;
#pragma unroll
    for (int i = 0; i < 16; ++i) {
        int a = tid + i * 256;
        float x = s[a];
        x = m[a] ? x : NEGBIG;
        vals[i] = x;
        lmax = fmaxf(lmax, x);
    }
#pragma unroll
    for (int off = 16; off >= 1; off >>= 1)
        lmax = fmaxf(lmax, __shfl_xor(lmax, off));
    if (lane == 0) redm[wave] = lmax;
    __syncthreads();
    float gmax = redm[0];
#pragma unroll
    for (int w = 1; w < 8; ++w) gmax = fmaxf(gmax, redm[w]);

    float lsum = 0.0f;
#pragma unroll
    for (int i = 0; i < 16; ++i) {
        vals[i] = __expf(vals[i] - gmax);
        lsum += vals[i];
    }
#pragma unroll
    for (int off = 16; off >= 1; off >>= 1)
        lsum += __shfl_xor(lsum, off);
    if (lane == 0) reds[wave] = lsum;
    __syncthreads();
    float gsum = 0.0f;
#pragma unroll
    for (int w = 0; w < 8; ++w) gsum += reds[w];

    const float inv = 1.0f / gsum;
    float* o = attn + (size_t)b * A_;
#pragma unroll
    for (int i = 0; i < 16; ++i)
        o[tid + i * 256] = vals[i] * inv;
}

// ---------------------------------------------------------------------------
// 5) partial[b][chunk][f] = sum_{a in chunk} attn[b][a] * inputs[b][a][f]
//    8 chunks of 512 rows; block = 512 threads (one per f), coalesced rows.
// ---------------------------------------------------------------------------
__global__ __launch_bounds__(512) void k_wsum(const float* __restrict__ inputs,
                                              const float* __restrict__ attn,
                                              float* __restrict__ partial) {
    __shared__ float aw[512];
    const int b     = blockIdx.y;
    const int chunk = blockIdx.x;
    const int f     = threadIdx.x;
    const int abase = chunk * 512;

    aw[f] = attn[(size_t)b * A_ + abase + f];
    __syncthreads();

    const float* base = inputs + ((size_t)b * A_ + abase) * F_ + f;
    float acc = 0.0f;
#pragma unroll 4
    for (int r = 0; r < 512; ++r)
        acc += aw[r] * base[(size_t)r * F_];

    partial[((size_t)b * 8 + chunk) * F_ + f] = acc;
}

// ---------------------------------------------------------------------------
// 6) out[b][f] = sum_chunk partial[b][chunk][f]
// ---------------------------------------------------------------------------
__global__ __launch_bounds__(512) void k_reduce(const float* __restrict__ partial,
                                                float* __restrict__ out) {
    const int b = blockIdx.x;
    const int f = threadIdx.x;
    float acc = 0.0f;
#pragma unroll
    for (int c = 0; c < 8; ++c)
        acc += partial[((size_t)b * 8 + c) * F_ + f];
    out[(size_t)b * F_ + f] = acc;
}

// ---------------------------------------------------------------------------
extern "C" void kernel_launch(void* const* d_in, const int* in_sizes, int n_in,
                              void* d_out, int out_size, void* d_ws, size_t ws_size,
                              hipStream_t stream) {
    const float*         inputs = (const float*)d_in[0];   // [B,A,F]
    const float*         ctx    = (const float*)d_in[1];   // [B,C]
    const unsigned char* mask   = (const unsigned char*)d_in[2]; // [B,A,1] bool
    const float*         W_ih   = (const float*)d_in[3];   // [H,F]
    const float*         W_ch   = (const float*)d_in[4];   // [H,C]
    const float*         W_s    = (const float*)d_in[5];   // [1,H]
    float*               out    = (float*)d_out;           // [B,F]

    char* ws = (char*)d_ws;
    unsigned short* wb      = (unsigned short*)(ws);                 // 512 KB
    float*          v       = (float*)(ws + (512u << 10));          //  64 KB
    float*          scores  = (float*)(ws + (576u << 10));          // 512 KB
    float*          attn    = (float*)(ws + (1088u << 10));         // 512 KB
    float*          partial = (float*)(ws + (1600u << 10));         // 512 KB
    (void)in_sizes; (void)n_in; (void)out_size; (void)ws_size;

    // 1) weights -> bf16
    k_cvt_w<<<(H_ * F_) / 256, 256, 0, stream>>>(W_ih, wb);
    // 2) fold context & W_s into v[b][h]
    k_ctx<<<B_, 512, 0, stream>>>(ctx, W_ch, W_s, v);
    // 3) WMMA scores pass (reads inputs once; async double-buffered weights)
    k_scores<<<(B_ * A_ / 16) / 8, 256, 0, stream>>>(inputs, wb, v, scores);
    // 4) masked softmax over A
    k_softmax<<<B_, 256, 0, stream>>>(scores, mask, attn);
    // 5) attn^T @ inputs partials (reads inputs once more)
    k_wsum<<<dim3(8, B_), 512, 0, stream>>>(inputs, attn, partial);
    // 6) final reduce
    k_reduce<<<B_, 512, 0, stream>>>(partial, out);
}
